// MPNN_75849122447742
// MI455X (gfx1250) — matrix operand order repaired and checked by reading, not compiled, not run on hardware
//
#include <hip/hip_runtime.h>
#include <hip/hip_bf16.h>

// MPNN fused implementation for gfx1250 (MI455X).
// Algebraic optimization: never materialize the E x 32 x 32 per-edge matrices.
// Precompute per-node U_n[j,k] = sum_h h_n[h] * e2w[j,h,k] (N << E), then
// msg_e = e_e @ U_src + h_src @ e2b_mat.
// Dense GEMMs use v_wmma_f32_16x16x32_f16 with pre-packed f16 B-fragments.

typedef __attribute__((ext_vector_type(16))) _Float16 v16h;
typedef __attribute__((ext_vector_type(8)))  float    v8f;

__device__ __forceinline__ v8f wmma_f16(v16h a, v16h b, v8f c) {
    // (neg_a, A, neg_b, B, c_mod, C, reuse_a, reuse_b)
    return __builtin_amdgcn_wmma_f32_16x16x32_f16(false, a, false, b,
                                                  (short)0, c, false, false);
}

// ---------------------------------------------------------------------------
// A fragment: 16xK f16 (K = 16 zero-padded to 32, or 32), rows row0..row0+15 of
// a row-major f32 matrix with lda == K. ISA wave32 layout:
//   lanes 0-15  -> row = lane,    elems 0..7 = K 0..7,  elems 8..15 = K 16..23
//   lanes 16-31 -> row = lane-16, elems 0..7 = K 8..15, elems 8..15 = K 24..31
// Loaded as contiguous float4s (rows are 16B-aligned) -> global_load_b128.
template <int K>
__device__ __forceinline__ v16h load_a(const float* __restrict__ A, int row0, int lane) {
    const float4* p4 = (const float4*)(A + (size_t)(row0 + (lane & 15)) * K);
    const int q = (lane & 16) ? 2 : 0;              // kh/4, kh in {0,8}
    const float4 x0 = p4[q];
    const float4 x1 = p4[q + 1];
    float4 x2 = {0.f, 0.f, 0.f, 0.f};
    float4 x3 = {0.f, 0.f, 0.f, 0.f};
    if (K == 32) { x2 = p4[q + 4]; x3 = p4[q + 5]; }
    v16h a;
    a[0] = (_Float16)x0.x; a[1] = (_Float16)x0.y; a[2] = (_Float16)x0.z; a[3] = (_Float16)x0.w;
    a[4] = (_Float16)x1.x; a[5] = (_Float16)x1.y; a[6] = (_Float16)x1.z; a[7] = (_Float16)x1.w;
    a[8]  = (_Float16)x2.x; a[9]  = (_Float16)x2.y; a[10] = (_Float16)x2.z; a[11] = (_Float16)x2.w;
    a[12] = (_Float16)x3.x; a[13] = (_Float16)x3.y; a[14] = (_Float16)x3.z; a[15] = (_Float16)x3.w;
    return a;
}

// ---------------------------------------------------------------------------
// Weight packing: convert a K x 32 row-major f32 weight matrix into the WMMA
// 32x16 B-fragment layout (f16), one v16h per (coltile, lane):
//   col = t*16 + (lane&15); elem i -> K = ((lane&16)?8:0) + i + (i>=8?8:0)  (zero if >= K).
// Packed matrix = 64 v16h entries (2 KB). Hot loops then load each fragment
// with a single 32-byte vector load instead of 16 strided scalar loads.
__global__ void pack_w_kernel(const float* __restrict__ W, int Kvalid,
                              v16h* __restrict__ out) {
    const int lane = threadIdx.x & 31;
    const int t    = threadIdx.x >> 5;              // blockDim = 64 -> t in {0,1}
    const int c    = t * 16 + (lane & 15);
    const int k0   = (lane & 16) ? 8 : 0;
    v16h b;
#pragma unroll
    for (int i = 0; i < 16; ++i) {
        const int k = k0 + i + (i >= 8 ? 8 : 0);
        b[i] = (_Float16)(k < Kvalid ? W[k * 32 + c] : 0.0f);
    }
    out[t * 32 + lane] = b;
}

// Pack the layer's e2w (32 slices of 32x32) plus e2b (reshaped 32x32) as 33
// packed matrices. grid = 33 blocks of 64 threads.
__global__ void pack_e2_kernel(const float* __restrict__ e2w,
                               const float* __restrict__ e2b,
                               v16h* __restrict__ out) {
    const int m    = blockIdx.x;                    // 0..31 = e2w slice j, 32 = e2b
    const float* W = (m < 32) ? (e2w + (size_t)m * 1024) : e2b;
    const int lane = threadIdx.x & 31;
    const int t    = threadIdx.x >> 5;
    const int c    = t * 16 + (lane & 15);
    const int k0   = (lane & 16) ? 8 : 0;
    v16h b;
#pragma unroll
    for (int i = 0; i < 16; ++i) b[i] = (_Float16)W[(k0 + i + (i >= 8 ? 8 : 0)) * 32 + c];
    out[(size_t)m * 64 + t * 32 + lane] = b;
}

// ---------------------------------------------------------------------------
// C[M x 32] = act(A[M x K] @ W[K x 32] + bias). One wave = one 16-row tile
// (two 16x16 WMMAs). STATS: accumulate per-column sum / sumsq for batchnorm.
template <int K, bool RELU, bool STATS>
__global__ __launch_bounds__(256) void gemm32_kernel(
        const float* __restrict__ A, const v16h* __restrict__ Wp,
        const float* __restrict__ bias, float* __restrict__ C,
        int M, float* __restrict__ stats) {
    const int lane = threadIdx.x & 31;
    const int wave = threadIdx.x >> 5;
    const int tile = blockIdx.x * (blockDim.x >> 5) + wave;
    const int row0 = tile * 16;
    if (row0 >= M) return;                          // uniform per wave: EXEC all-1s

    const v16h a  = load_a<K>(A, row0, lane);
    const v16h b0 = Wp[lane];
    const v16h b1 = Wp[32 + lane];
    v8f c0 = {}, c1 = {};
    c0 = wmma_f16(a, b0, c0);
    c1 = wmma_f16(a, b1, c1);

    const int n0 = lane & 15;
    const int mb = row0 + ((lane & 16) ? 8 : 0);
    const float bz0 = bias[n0];
    const float bz1 = bias[16 + n0];
    float s0 = 0.f, q0 = 0.f, s1 = 0.f, q1 = 0.f;
#pragma unroll
    for (int r = 0; r < 8; ++r) {
        float v0 = c0[r] + bz0;
        float v1 = c1[r] + bz1;
        if (RELU) { v0 = fmaxf(v0, 0.f); v1 = fmaxf(v1, 0.f); }
        C[(size_t)(mb + r) * 32 + n0] = v0;
        C[(size_t)(mb + r) * 32 + 16 + n0] = v1;
        if (STATS) { s0 += v0; q0 += v0 * v0; s1 += v1; q1 += v1 * v1; }
    }
    if (STATS) {
        atomicAdd(&stats[n0],      s0);
        atomicAdd(&stats[16 + n0], s1);
        atomicAdd(&stats[32 + n0], q0);
        atomicAdd(&stats[48 + n0], q1);
    }
}

// ---------------------------------------------------------------------------
// Per-node factor: U[n][j][k] = sum_h h[n][h] * e2w[j][h*32+k]  (32 GEMMs,
// shared A fragment), and Bv[n][k] = sum_h h[n][h] * e2b[h*32+k].
// Block = one 16-node tile, 8 waves; wave w handles j = w, w+8, w+16, w+24.
// e2p = packed fragments for the 33 matrices (e2w slices + e2b).
__global__ __launch_bounds__(256) void ufactor_kernel(
        const float* __restrict__ h, const v16h* __restrict__ e2p,
        float* __restrict__ U, float* __restrict__ Bv, int N) {
    const int lane = threadIdx.x & 31;
    const int wave = threadIdx.x >> 5;
    const int row0 = blockIdx.x * 16;
    if (row0 >= N) return;

    const v16h a = load_a<32>(h, row0, lane);
    const int n0 = lane & 15;
    const int mb = row0 + ((lane & 16) ? 8 : 0);

#pragma unroll
    for (int jj = 0; jj < 4; ++jj) {
        const int j = wave + jj * 8;
        const v16h b0 = e2p[(size_t)j * 64 + lane];
        const v16h b1 = e2p[(size_t)j * 64 + 32 + lane];
        v8f c0 = {}, c1 = {};
        c0 = wmma_f16(a, b0, c0);
        c1 = wmma_f16(a, b1, c1);
        float* Ud = U + (size_t)j * 32;
#pragma unroll
        for (int r = 0; r < 8; ++r) {
            Ud[(size_t)(mb + r) * 1024 + n0]      = c0[r];
            Ud[(size_t)(mb + r) * 1024 + 16 + n0] = c1[r];
        }
    }
    if (wave == 0) {   // bias-term matrix: Bv = h @ reshape(e2b, 32x32)
        const v16h b0 = e2p[(size_t)32 * 64 + lane];
        const v16h b1 = e2p[(size_t)32 * 64 + 32 + lane];
        v8f c0 = {}, c1 = {};
        c0 = wmma_f16(a, b0, c0);
        c1 = wmma_f16(a, b1, c1);
#pragma unroll
        for (int r = 0; r < 8; ++r) {
            Bv[(size_t)(mb + r) * 32 + n0]      = c0[r];
            Bv[(size_t)(mb + r) * 32 + 16 + n0] = c1[r];
        }
    }
}

// ---------------------------------------------------------------------------
// Per-edge message + scatter: one wave per edge, lane = output feature k.
// msg_e[k] = Bv[src][k] + sum_j e[e][j] * U[src][j][k];  atomicAdd into agg[dst].
__global__ __launch_bounds__(256) void msg_kernel(
        const float* __restrict__ ebuf, const float* __restrict__ U,
        const float* __restrict__ Bv, const int* __restrict__ edge_index,
        float* __restrict__ agg, int E) {
    const int ew = (int)((blockIdx.x * blockDim.x + threadIdx.x) >> 5);
    const int lane = threadIdx.x & 31;
    if (ew >= E) return;
    const int s = edge_index[ew];        // row 0 of (2,E): src
    const int d = edge_index[E + ew];    // row 1: dst
    const float ej = ebuf[(size_t)ew * 32 + lane];
    const float* Ue = U + (size_t)s * 1024;
    float acc = Bv[(size_t)s * 32 + lane];
#pragma unroll
    for (int j = 0; j < 32; ++j) {
        const float e_j = __shfl(ej, j, 32);
        acc = fmaf(e_j, Ue[(size_t)j * 32 + lane], acc);
    }
    atomicAdd(&agg[(size_t)d * 32 + lane], acc);
}

// ---------------------------------------------------------------------------
__global__ void zero_kernel(float* __restrict__ p, int n) {
    const int t = blockIdx.x * blockDim.x + threadIdx.x;
    if (t < n) p[t] = 0.0f;
}

// BatchNorm apply (in place) + relu. stats: [0..31]=sum, [32..63]=sumsq.
__global__ void bn_kernel(float* __restrict__ h, const float* __restrict__ stats,
                          const float* __restrict__ gamma,
                          const float* __restrict__ beta, int N) {
    const int t = blockIdx.x * blockDim.x + threadIdx.x;
    if (t >= N * 32) return;
    const int c = t & 31;
    const float invN = 1.0f / (float)N;
    const float mean = stats[c] * invN;
    const float var  = stats[32 + c] * invN - mean * mean;
    const float v = (h[t] - mean) * rsqrtf(var + 1e-5f) * gamma[c] + beta[c];
    h[t] = v > 0.f ? v : 0.f;
}

// Segment sum into per-graph buckets + counts.
__global__ void pool_kernel(const float* __restrict__ h, const int* __restrict__ batch,
                            float* __restrict__ gsums, float* __restrict__ gcnt, int N) {
    const int t = blockIdx.x * blockDim.x + threadIdx.x;
    if (t >= N * 32) return;
    const int n = t >> 5, c = t & 31;
    const int g = batch[n];
    atomicAdd(&gsums[(size_t)g * 32 + c], h[t]);
    if (c == 0) atomicAdd(&gcnt[g], 1.0f);
}

// z[g][j] = relu(mean-pooled[g] @ p1w + p1b), G=128, 16 outputs
__global__ void mlp1_kernel(const float* __restrict__ gsums, const float* __restrict__ gcnt,
                            const float* __restrict__ p1w, const float* __restrict__ p1b,
                            float* __restrict__ z) {
    const int t = blockIdx.x * blockDim.x + threadIdx.x;
    if (t >= 128 * 16) return;
    const int g = t >> 4, j = t & 15;
    float cnt = gcnt[g]; cnt = cnt > 1.0f ? cnt : 1.0f;
    const float inv = 1.0f / cnt;
    float acc = p1b[j];
#pragma unroll
    for (int c = 0; c < 32; ++c)
        acc = fmaf(gsums[(size_t)g * 32 + c] * inv, p1w[c * 16 + j], acc);
    z[t] = acc > 0.f ? acc : 0.f;
}

__global__ void mlp2_kernel(const float* __restrict__ z, const float* __restrict__ p2w,
                            const float* __restrict__ p2b, float* __restrict__ out) {
    const int t = blockIdx.x * blockDim.x + threadIdx.x;
    if (t >= 128 * 16) return;
    const int g = t >> 4, o = t & 15;
    float acc = p2b[o];
#pragma unroll
    for (int j = 0; j < 16; ++j)
        acc = fmaf(z[g * 16 + j], p2w[j * 16 + o], acc);
    out[t] = acc;
}

// ---------------------------------------------------------------------------
extern "C" void kernel_launch(void* const* d_in, const int* in_sizes, int n_in,
                              void* d_out, int out_size, void* d_ws, size_t ws_size,
                              hipStream_t stream) {
    (void)in_sizes; (void)n_in; (void)out_size; (void)ws_size;
    const int N = 10000, E = 160000, G = 128;

    const float* x          = (const float*)d_in[0];
    const int*   edge_index = (const int*)  d_in[1];
    const float* edge_attr  = (const float*)d_in[2];
    const int*   batch      = (const int*)  d_in[3];
    // params flattened in jax pytree (sorted-key) order:
    // embed_b, embed_w, layers[0..2]{beta,e1b,e1w,e2b,e2w,gamma,nb,nw}, p1b,p1w,p2b,p2w
    const float* embed_b = (const float*)d_in[4];
    const float* embed_w = (const float*)d_in[5];
    struct Layer { const float *beta, *e1b, *e1w, *e2b, *e2w, *gamma, *nb, *nw; };
    Layer L[3];
    for (int l = 0; l < 3; ++l) {
        const int base = 6 + 8 * l;
        L[l].beta  = (const float*)d_in[base + 0];
        L[l].e1b   = (const float*)d_in[base + 1];
        L[l].e1w   = (const float*)d_in[base + 2];
        L[l].e2b   = (const float*)d_in[base + 3];
        L[l].e2w   = (const float*)d_in[base + 4];
        L[l].gamma = (const float*)d_in[base + 5];
        L[l].nb    = (const float*)d_in[base + 6];
        L[l].nw    = (const float*)d_in[base + 7];
    }
    const float* p1b = (const float*)d_in[30];
    const float* p1w = (const float*)d_in[31];
    const float* p2b = (const float*)d_in[32];
    const float* p2w = (const float*)d_in[33];

    // Workspace carve-up (~67 MB activations + ~220 KB packed weights)
    float* ws = (float*)d_ws;
    float* hA    = ws; ws += (size_t)N * 32;
    float* hB    = ws; ws += (size_t)N * 32;
    float* ebuf  = ws; ws += (size_t)E * 32;
    float* U     = ws; ws += (size_t)N * 1024;
    float* Bv    = ws; ws += (size_t)N * 32;
    float* agg   = ws; ws += (size_t)N * 32;   // stats immediately after -> one zero pass
    float* stats = ws; ws += 64;
    float* gbuf  = ws; ws += (size_t)G * 33;   // G*32 sums + G counts
    float* zbuf  = ws; ws += (size_t)G * 16;
    // packed weight fragments (v16h = 32 B each); ws is 64B-aligned here
    v16h* pk       = (v16h*)ws;
    v16h* pk_embed = pk;                pk += 64;
    v16h* pk_e1[3]; v16h* pk_nw[3]; v16h* pk_e2[3];
    for (int l = 0; l < 3; ++l) { pk_e1[l] = pk; pk += 64; }
    for (int l = 0; l < 3; ++l) { pk_nw[l] = pk; pk += 64; }
    for (int l = 0; l < 3; ++l) { pk_e2[l] = pk; pk += (size_t)33 * 64; }

    const dim3 blk(256);
    const int tilesN = N / 16;   // 625
    const int tilesE = E / 16;   // 10000

    // Pack all weights into WMMA B-fragment layout (tiny, 10 launches)
    pack_w_kernel<<<1, 64, 0, stream>>>(embed_w, 32, pk_embed);
    for (int l = 0; l < 3; ++l) {
        pack_w_kernel<<<1, 64, 0, stream>>>(L[l].e1w, 16, pk_e1[l]);
        pack_w_kernel<<<1, 64, 0, stream>>>(L[l].nw, 32, pk_nw[l]);
        pack_e2_kernel<<<33, 64, 0, stream>>>(L[l].e2w, L[l].e2b, pk_e2[l]);
    }

    // h = x @ embed_w + embed_b
    gemm32_kernel<32, false, false><<<(tilesN + 7) / 8, blk, 0, stream>>>(
        x, pk_embed, embed_b, hA, N, nullptr);

    float* hc = hA;
    float* hn = hB;
    for (int l = 0; l < 3; ++l) {
        // e = relu(edge_attr @ e1w + e1b)   (K=16 zero-padded to 32)
        gemm32_kernel<16, true, false><<<(tilesE + 7) / 8, blk, 0, stream>>>(
            edge_attr, pk_e1[l], L[l].e1b, ebuf, E, nullptr);
        // U_n = h_n contracted with e2w over the middle index; Bv = h @ e2b_mat
        ufactor_kernel<<<tilesN, blk, 0, stream>>>(hc, pk_e2[l], U, Bv, N);
        // zero agg + stats in one pass
        zero_kernel<<<(N * 32 + 64 + 255) / 256, blk, 0, stream>>>(agg, N * 32 + 64);
        // msg_e = e_e @ U_src + Bv_src, scattered to agg[dst]
        msg_kernel<<<E / 8, blk, 0, stream>>>(ebuf, U, Bv, edge_index, agg, E);
        // h' = relu(agg @ nw + nb), fused BN stats accumulation
        gemm32_kernel<32, true, true><<<(tilesN + 7) / 8, blk, 0, stream>>>(
            agg, pk_nw[l], L[l].nb, hn, N, stats);
        // batchnorm + relu in place
        bn_kernel<<<(N * 32 + 255) / 256, blk, 0, stream>>>(hn, stats, L[l].gamma,
                                                            L[l].beta, N);
        float* tmp = hc; hc = hn; hn = tmp;
    }

    // mean pool per graph, then 2-layer head
    zero_kernel<<<(G * 33 + 255) / 256, blk, 0, stream>>>(gbuf, G * 33);
    pool_kernel<<<(N * 32 + 255) / 256, blk, 0, stream>>>(hc, batch, gbuf,
                                                          gbuf + (size_t)G * 32, N);
    mlp1_kernel<<<(G * 16 + 255) / 256, blk, 0, stream>>>(gbuf, gbuf + (size_t)G * 32,
                                                          p1w, p1b, zbuf);
    mlp2_kernel<<<(G * 16 + 255) / 256, blk, 0, stream>>>(zbuf, p2w, p2b, (float*)d_out);
}
